// RNNBase_3968549782372
// MI455X (gfx1250) — 8-bit matrix operand order repaired and checked by reading, compiled, not run on hardware
//
#include <hip/hip_runtime.h>
#include <cstdint>


typedef __attribute__((ext_vector_type(8))) int v8i;

constexpr int kT    = 512;
constexpr int kB    = 32;
constexpr int kH    = 1024;
constexpr int kG    = 4096;
constexpr int kWN   = kG * kH;
constexpr int kActN = kB * kH;

__device__ __forceinline__ v8i load_a_frag(const int8_t* base, int m0, int k0,
                                           int lane) {
  const int row  = m0 + (lane & 15);
  const int koff = (lane < 16) ? 0 : 16;
  const int8_t* p = base + (size_t)row * kH + k0 + koff;
  v8i a;
#pragma unroll
  for (int j = 0; j < 4; ++j) {
    const int* q = (const int*)(p + 32 * (j >> 1) + 8 * (j & 1));
    a[2 * j]     = q[0];
    a[2 * j + 1] = q[1];
  }
  return a;
}

__device__ __forceinline__ v8i load_b_frag(const int8_t* w, int g0, int k0,
                                           int lane) {
  const int n    = lane & 15;
  const int koff = (lane < 16) ? 0 : 16;
  const int8_t* p = w + (size_t)(g0 + n) * kH + k0 + koff;
  const int* q0 = (const int*)p;
  const int* q1 = (const int*)(p + 32);
  v8i b;
#pragma unroll
  for (int j = 0; j < 4; ++j) {
    b[j]     = q0[j];
    b[4 + j] = q1[j];
  }
  return b;
}

__global__ __launch_bounds__(256) void k_init(const float* __restrict__ c0,
                                              float* __restrict__ cS,
                                              unsigned* __restrict__ wmax,
                                              unsigned* __restrict__ hAbs0,
                                              unsigned* __restrict__ hAbs1) {
  const int i = blockIdx.x * blockDim.x + threadIdx.x;
  const int stride = gridDim.x * blockDim.x;
  if (i < 4) wmax[i] = 0u;
  for (int j = i; j <= kT; j += stride) {
    hAbs0[j] = 0u;
    hAbs1[j] = 0u;
  }
  for (int j = i; j < 2 * kActN; j += stride) cS[j] = c0[j];
}

__global__ __launch_bounds__(256) void k_h0abs(const float* __restrict__ h0,
                                               unsigned* __restrict__ hAbs0,
                                               unsigned* __restrict__ hAbs1) {
  const float* s = h0 + (size_t)blockIdx.x * kActN;
  float m = 0.f;
  for (int i = threadIdx.x; i < kActN; i += 256) m = fmaxf(m, fabsf(s[i]));
  __shared__ float red[256];
  red[threadIdx.x] = m;
  __syncthreads();
  for (int r = 128; r > 0; r >>= 1) {
    if ((int)threadIdx.x < r)
      red[threadIdx.x] = fmaxf(red[threadIdx.x], red[threadIdx.x + r]);
    __syncthreads();
  }
  if (threadIdx.x == 0) {
    unsigned* slot = (blockIdx.x == 0) ? hAbs0 : hAbs1;
    slot[0] = __float_as_uint(red[0]);
  }
}

__global__ __launch_bounds__(256) void k_wabsmax(const float* __restrict__ w0,
                                                 const float* __restrict__ w1,
                                                 const float* __restrict__ w2,
                                                 const float* __restrict__ w3,
                                                 unsigned* __restrict__ wmax) {
  const float* w = (blockIdx.y == 0) ? w0
                 : (blockIdx.y == 1) ? w1
                 : (blockIdx.y == 2) ? w2 : w3;
  float m = 0.f;
  for (int i = blockIdx.x * blockDim.x + threadIdx.x; i < kWN;
       i += gridDim.x * blockDim.x)
    m = fmaxf(m, fabsf(w[i]));
  __shared__ float red[256];
  red[threadIdx.x] = m;
  __syncthreads();
  for (int s = 128; s > 0; s >>= 1) {
    if ((int)threadIdx.x < s)
      red[threadIdx.x] = fmaxf(red[threadIdx.x], red[threadIdx.x + s]);
    __syncthreads();
  }
  if (threadIdx.x == 0) atomicMax(&wmax[blockIdx.y], __float_as_uint(red[0]));
}

__global__ __launch_bounds__(256) void k_wquant(const float* __restrict__ w0,
                                                const float* __restrict__ w1,
                                                const float* __restrict__ w2,
                                                const float* __restrict__ w3,
                                                const unsigned* __restrict__ wmax,
                                                int8_t* __restrict__ q0,
                                                int8_t* __restrict__ q1,
                                                int8_t* __restrict__ q2,
                                                int8_t* __restrict__ q3) {
  const int y = blockIdx.y;
  const float* w = (y == 0) ? w0 : (y == 1) ? w1 : (y == 2) ? w2 : w3;
  int8_t* q      = (y == 0) ? q0 : (y == 1) ? q1 : (y == 2) ? q2 : q3;
  const float scale = fmaxf(__uint_as_float(wmax[y]), 1e-8f) * (1.f / 127.f);
  const float inv   = 1.f / scale;
  for (int i = blockIdx.x * blockDim.x + threadIdx.x; i < kWN;
       i += gridDim.x * blockDim.x) {
    float v = fminf(fmaxf(w[i] * inv, -127.f), 127.f);
    q[i] = (int8_t)__float2int_rn(v);
  }
}

__global__ __launch_bounds__(256) void k_actquant(const float* __restrict__ src,
                                                  int8_t* __restrict__ dst,
                                                  float* __restrict__ scaleOut,
                                                  int perBlock) {
  const float* s = src + (size_t)blockIdx.x * perBlock;
  int8_t* d      = dst + (size_t)blockIdx.x * perBlock;
  float m = 0.f;
  for (int i = threadIdx.x; i < perBlock; i += 256) m = fmaxf(m, fabsf(s[i]));
  __shared__ float red[256];
  red[threadIdx.x] = m;
  __syncthreads();
  for (int r = 128; r > 0; r >>= 1) {
    if ((int)threadIdx.x < r)
      red[threadIdx.x] = fmaxf(red[threadIdx.x], red[threadIdx.x + r]);
    __syncthreads();
  }
  const float scale = fmaxf(red[0], 1e-8f) * (1.f / 127.f);
  if (threadIdx.x == 0) scaleOut[blockIdx.x] = scale;
  const float inv = 1.f / scale;
  for (int i = threadIdx.x; i < perBlock; i += 256) {
    float v = fminf(fmaxf(s[i] * inv, -127.f), 127.f);
    d[i] = (int8_t)__float2int_rn(v);
  }
}

__global__ __launch_bounds__(128) void k_lstm_cell(
    const int8_t* __restrict__ qx_t, const float* __restrict__ sx_t,
    const float* __restrict__ hIn, const unsigned* __restrict__ hAbsT,
    unsigned* __restrict__ hAbsNext,
    const int8_t* __restrict__ qWi, const unsigned* __restrict__ wmaxI,
    const int8_t* __restrict__ qWh, const unsigned* __restrict__ wmaxH,
    const float* __restrict__ b_ih, const float* __restrict__ b_hh,
    float* __restrict__ cS, float* __restrict__ hOut) {
  __shared__ int8_t qhL[kB * kH];
  __shared__ float gbuf[4][kB][16];
  __shared__ float redMax[128];

  const int lane = threadIdx.x & 31;
  const int gate = threadIdx.x >> 5;
  const int cb   = blockIdx.x;
  const int g0   = gate * kH + cb * 16;

  const float shScale = fmaxf(__uint_as_float(hAbsT[0]), 1e-8f) * (1.f / 127.f);
  const float hInv = 1.f / shScale;
  for (int i = threadIdx.x; i < kB * kH; i += 128) {
    float v = fminf(fmaxf(hIn[i] * hInv, -127.f), 127.f);
    qhL[i] = (int8_t)__float2int_rn(v);
  }
  __syncthreads();

  v8i accX0 = {}, accX1 = {}, accH0 = {}, accH1 = {};
#pragma unroll 4
  for (int k0 = 0; k0 < kH; k0 += 64) {
    v8i bI = load_b_frag(qWi, g0, k0, lane);
    v8i x0 = load_a_frag(qx_t, 0, k0, lane);
    v8i x1 = load_a_frag(qx_t, 16, k0, lane);
    accX0 = __builtin_amdgcn_wmma_i32_16x16x64_iu8(true, x0, true, bI, accX0,
                                                   false, false);
    accX1 = __builtin_amdgcn_wmma_i32_16x16x64_iu8(true, x1, true, bI, accX1,
                                                   false, false);
    v8i bH = load_b_frag(qWh, g0, k0, lane);
    v8i h0f = load_a_frag(qhL, 0, k0, lane);
    v8i h1f = load_a_frag(qhL, 16, k0, lane);
    accH0 = __builtin_amdgcn_wmma_i32_16x16x64_iu8(true, h0f, true, bH, accH0,
                                                   false, false);
    accH1 = __builtin_amdgcn_wmma_i32_16x16x64_iu8(true, h1f, true, bH, accH1,
                                                   false, false);
  }

  const float si  = fmaxf(__uint_as_float(wmaxI[0]), 1e-8f) * (1.f / 127.f);
  const float shw = fmaxf(__uint_as_float(wmaxH[0]), 1e-8f) * (1.f / 127.f);
  const float fI  = sx_t[0] * si;
  const float fH  = shScale * shw;

  const int col  = lane & 15;
  const int rofs = (lane < 16) ? 0 : 8;
  const float bias = b_ih[g0 + col] + b_hh[g0 + col];
#pragma unroll
  for (int i = 0; i < 8; ++i) {
    gbuf[gate][rofs + i][col] =
        (float)accX0[i] * fI + (float)accH0[i] * fH + bias;
    gbuf[gate][16 + rofs + i][col] =
        (float)accX1[i] * fI + (float)accH1[i] * fH + bias;
  }
  __syncthreads();

  float lmax = 0.f;
  for (int idx = threadIdx.x; idx < kB * 16; idx += 128) {
    const int b  = idx >> 4;
    const int n  = idx & 15;
    const int ch = cb * 16 + n;
    const float ig = 1.f / (1.f + expf(-gbuf[0][b][n]));
    const float fg = 1.f / (1.f + expf(-gbuf[1][b][n]));
    const float gg = tanhf(gbuf[2][b][n]);
    const float og = 1.f / (1.f + expf(-gbuf[3][b][n]));
    const size_t off = (size_t)b * kH + ch;
    const float c = fg * cS[off] + ig * gg;
    const float h = og * tanhf(c);
    cS[off]   = c;
    hOut[off] = h;
    lmax = fmaxf(lmax, fabsf(h));
  }

  redMax[threadIdx.x] = lmax;
  __syncthreads();
  for (int r = 64; r > 0; r >>= 1) {
    if ((int)threadIdx.x < r)
      redMax[threadIdx.x] = fmaxf(redMax[threadIdx.x], redMax[threadIdx.x + r]);
    __syncthreads();
  }
  if (threadIdx.x == 0) atomicMax(hAbsNext, __float_as_uint(redMax[0]));
}

__global__ __launch_bounds__(256) void k_finalize(const float* __restrict__ y0Last,
                                                  const float* __restrict__ y1Last,
                                                  const float* __restrict__ cS,
                                                  float* __restrict__ outTail) {
  const int i = blockIdx.x * blockDim.x + threadIdx.x;
  if (i < kActN) {
    outTail[i]             = y0Last[i];
    outTail[kActN + i]     = y1Last[i];
    outTail[2 * kActN + i] = cS[i];
    outTail[3 * kActN + i] = cS[kActN + i];
  }
}

extern "C" void kernel_launch(void* const* d_in, const int* in_sizes, int n_in,
                              void* d_out, int out_size, void* d_ws,
                              size_t ws_size, hipStream_t stream) {
  (void)in_sizes; (void)n_in; (void)out_size; (void)ws_size;
  const float* x     = (const float*)d_in[0];
  const float* h0    = (const float*)d_in[1];
  const float* c0    = (const float*)d_in[2];
  const float* w_ih0 = (const float*)d_in[3];
  const float* w_hh0 = (const float*)d_in[4];
  const float* b_ih0 = (const float*)d_in[5];
  const float* b_hh0 = (const float*)d_in[6];
  const float* w_ih1 = (const float*)d_in[7];
  const float* w_hh1 = (const float*)d_in[8];
  const float* b_ih1 = (const float*)d_in[9];
  const float* b_hh1 = (const float*)d_in[10];
  float* out = (float*)d_out;

  char* ws = (char*)d_ws;
  size_t off = 0;
  auto carve = [&](size_t bytes) -> void* {
    void* p = ws + off;
    off = (off + bytes + 255) & ~(size_t)255;
    return p;
  };
  int8_t* qWi0 = (int8_t*)carve(kWN);
  int8_t* qWh0 = (int8_t*)carve(kWN);
  int8_t* qWi1 = (int8_t*)carve(kWN);
  int8_t* qWh1 = (int8_t*)carve(kWN);
  int8_t* qx   = (int8_t*)carve((size_t)kT * kActN);
  float*  y0   = (float*)carve((size_t)kT * kActN * sizeof(float));
  float*  cS   = (float*)carve(2 * kActN * sizeof(float));
  float*  sxArr   = (float*)carve(kT * sizeof(float));
  unsigned* hAbs0 = (unsigned*)carve((kT + 1) * sizeof(unsigned));
  unsigned* hAbs1 = (unsigned*)carve((kT + 1) * sizeof(unsigned));
  unsigned* wmax  = (unsigned*)carve(4 * sizeof(unsigned));

  k_init<<<256, 256, 0, stream>>>(c0, cS, wmax, hAbs0, hAbs1);
  k_h0abs<<<2, 256, 0, stream>>>(h0, hAbs0, hAbs1);
  dim3 wgrid(128, 4);
  k_wabsmax<<<wgrid, 256, 0, stream>>>(w_ih0, w_hh0, w_ih1, w_hh1, wmax);
  k_wquant<<<wgrid, 256, 0, stream>>>(w_ih0, w_hh0, w_ih1, w_hh1, wmax,
                                      qWi0, qWh0, qWi1, qWh1);

  k_actquant<<<kT, 256, 0, stream>>>(x, qx, sxArr, kActN);
  for (int t = 0; t < kT; ++t) {
    const float* hIn = (t == 0) ? h0 : y0 + (size_t)(t - 1) * kActN;
    k_lstm_cell<<<64, 128, 0, stream>>>(qx + (size_t)t * kActN, sxArr + t, hIn,
                                        hAbs0 + t, hAbs0 + t + 1, qWi0,
                                        wmax + 0, qWh0, wmax + 1, b_ih0, b_hh0,
                                        cS, y0 + (size_t)t * kActN);
  }

  k_actquant<<<kT, 256, 0, stream>>>(y0, qx, sxArr, kActN);
  for (int t = 0; t < kT; ++t) {
    const float* hIn = (t == 0) ? h0 + kActN : out + (size_t)(t - 1) * kActN;
    k_lstm_cell<<<64, 128, 0, stream>>>(qx + (size_t)t * kActN, sxArr + t, hIn,
                                        hAbs1 + t, hAbs1 + t + 1, qWi1,
                                        wmax + 2, qWh1, wmax + 3, b_ih1, b_hh1,
                                        cS + kActN, out + (size_t)t * kActN);
  }

  k_finalize<<<128, 256, 0, stream>>>(y0 + (size_t)(kT - 1) * kActN,
                                      out + (size_t)(kT - 1) * kActN, cS,
                                      out + (size_t)kT * kActN);
}
